// Seq2Seq_58480274703182
// MI455X (gfx1250) — matrix operand order repaired and checked by reading, not compiled, not run on hardware
//
#include <hip/hip_runtime.h>
#include <hip/hip_bf16.h>

#define SS  32      // source length
#define TT  32      // target length
#define BB  512     // batch
#define HH  1024    // hidden
#define EE  256     // embed
#define AAD 1024    // attention dim
#define VSZ 64      // src vocab
#define VTT 70      // tgt vocab
#define START_TOK 1

typedef _Float16 h16 __attribute__((ext_vector_type(16)));
typedef _Float16 h8v __attribute__((ext_vector_type(8)));
typedef float    f8v __attribute__((ext_vector_type(8)));
typedef float    f4v __attribute__((ext_vector_type(4)));

// ---------------------------------------------------------------------------
// WMMA GEMM:  C[M,N] = A[M,K] (f16) @ W[N,K]^T (f16) + bias[N]   (f32 accum)
// One wave computes a 32x64 tile of C: 2 A-frags x 4 B-frags = 8 WMMA / chunk.
// Requires M%32==0, N%64==0, K%32==0 (true for every call site here).
// grid = (N/64, M/32), block = 32.
// ---------------------------------------------------------------------------
__global__ __launch_bounds__(32) void gemm_wmma_f16(
    const _Float16* __restrict__ A, const _Float16* __restrict__ W,
    const float* __restrict__ bias, float* __restrict__ C,
    int K, int lda, int ldw, int ldc)
{
    const int lane = threadIdx.x & 31;
    const int m0   = blockIdx.y << 5;
    const int n0   = blockIdx.x << 6;
    const int ncol = lane & 15;
    const int grp  = lane >> 4;              // lane group 0/1

    const _Float16* __restrict__ Arow0 = A + (size_t)(m0 + ncol) * lda;
    const _Float16* __restrict__ Arow1 = Arow0 + (size_t)16 * lda;
    const _Float16* __restrict__ Wb    = W + (size_t)(n0 + ncol) * ldw + (size_t)grp * 16;

    f8v acc[2][4] = {};
    for (int k0 = 0; k0 < K; k0 += 32) {
        // A fragments (contiguous K order, matching B): elems 0..7 = K[grp*16..+7],
        // elems 8..15 = K[grp*16+8..+7]
        const int ka = k0 + grp * 16;
        h8v a0lo = *(const h8v*)(Arow0 + ka);
        h8v a0hi = *(const h8v*)(Arow0 + ka + 8);
        h8v a1lo = *(const h8v*)(Arow1 + ka);
        h8v a1hi = *(const h8v*)(Arow1 + ka + 8);
        h16 af0, af1;
#pragma unroll
        for (int i = 0; i < 8; ++i) {
            af0[i] = a0lo[i]; af0[8 + i] = a0hi[i];
            af1[i] = a1lo[i]; af1[8 + i] = a1hi[i];
        }
        // B fragments: lane holds 16 contiguous K-values of column n
        const _Float16* Wp = Wb + k0;
        h16 b0 = *(const h16*)(Wp);
        h16 b1 = *(const h16*)(Wp + (size_t)16 * ldw);
        h16 b2 = *(const h16*)(Wp + (size_t)32 * ldw);
        h16 b3 = *(const h16*)(Wp + (size_t)48 * ldw);

        acc[0][0] = __builtin_amdgcn_wmma_f32_16x16x32_f16(false, af0, false, b0, (short)0, acc[0][0], false, false);
        acc[0][1] = __builtin_amdgcn_wmma_f32_16x16x32_f16(false, af0, false, b1, (short)0, acc[0][1], false, false);
        acc[0][2] = __builtin_amdgcn_wmma_f32_16x16x32_f16(false, af0, false, b2, (short)0, acc[0][2], false, false);
        acc[0][3] = __builtin_amdgcn_wmma_f32_16x16x32_f16(false, af0, false, b3, (short)0, acc[0][3], false, false);
        acc[1][0] = __builtin_amdgcn_wmma_f32_16x16x32_f16(false, af1, false, b0, (short)0, acc[1][0], false, false);
        acc[1][1] = __builtin_amdgcn_wmma_f32_16x16x32_f16(false, af1, false, b1, (short)0, acc[1][1], false, false);
        acc[1][2] = __builtin_amdgcn_wmma_f32_16x16x32_f16(false, af1, false, b2, (short)0, acc[1][2], false, false);
        acc[1][3] = __builtin_amdgcn_wmma_f32_16x16x32_f16(false, af1, false, b3, (short)0, acc[1][3], false, false);
    }
#pragma unroll
    for (int f = 0; f < 2; ++f) {
#pragma unroll
        for (int j = 0; j < 4; ++j) {
            const int n  = n0 + j * 16 + ncol;
            const float bv = bias ? bias[n] : 0.0f;
#pragma unroll
            for (int r = 0; r < 8; ++r) {
                const int m = m0 + f * 16 + grp * 8 + r;
                C[(size_t)m * ldc + n] = acc[f][j][r] + bv;
            }
        }
    }
}

// ---------------------------------------------------------------------------
// Helpers
// ---------------------------------------------------------------------------
__global__ void cvt_f16_kernel(const float* __restrict__ in, _Float16* __restrict__ out, int n)
{
    int i = blockIdx.x * 256 + threadIdx.x;
    if (i < n) out[i] = (_Float16)in[i];
}

// W_attn (2H, A) f32 -> W_attn_T (A, 2H) f16
__global__ void transpose_cvt_kernel(const float* __restrict__ in, _Float16* __restrict__ out)
{
    int i = blockIdx.x * 256 + threadIdx.x;       // over 2H*A
    if (i < 2 * HH * AAD) {
        int f = i / AAD, a = i % AAD;
        out[(size_t)a * (2 * HH) + f] = (_Float16)in[i];
    }
}

__global__ void zero_f32_kernel(float* __restrict__ p, int n)
{
    int i = blockIdx.x * 256 + threadIdx.x;
    if (i < n) p[i] = 0.0f;
}

__global__ void zero_f16_kernel(_Float16* __restrict__ p, int n)
{
    int i = blockIdx.x * 256 + threadIdx.x;
    if (i < n) p[i] = (_Float16)0.0f;
}

// outputs[0] = one-hot START row
__global__ void init_row0_kernel(float* __restrict__ out)
{
    int i = blockIdx.x * 256 + threadIdx.x;       // B*VT
    if (i < BB * VTT) out[i] = ((i % VTT) == START_TOK) ? 1.0f : 0.0f;
}

__global__ void init_xcur_kernel(const int* __restrict__ target0, int* __restrict__ xcur)
{
    int b = blockIdx.x * 256 + threadIdx.x;
    if (b < BB) xcur[b] = target0[b];
}

// gather one encoder step's embeddings, emit f16: (B, E)
__global__ void gather_src_kernel(const int* __restrict__ src_s, const float* __restrict__ emb,
                                  _Float16* __restrict__ out)
{
    int i = blockIdx.x * 256 + threadIdx.x;       // B*E
    if (i < BB * EE) {
        int b = i / EE, e = i % EE;
        out[i] = (_Float16)emb[(size_t)src_s[b] * EE + e];
    }
}

// torch GRUCell gate math; writes f32 h and an f16 shadow for the GEMMs
__global__ void gru_gate_kernel(const float* __restrict__ gi, const float* __restrict__ gh,
                                const float* __restrict__ h, float* __restrict__ hnew,
                                _Float16* __restrict__ hnew_h)
{
    int i = blockIdx.x * 256 + threadIdx.x;       // B*H
    if (i >= BB * HH) return;
    int b = i >> 10;                               // / H
    int j = i & (HH - 1);
    size_t base = (size_t)b * (3 * HH);
    float ir = gi[base + j], iz = gi[base + HH + j], in_ = gi[base + 2 * HH + j];
    float hr = gh[base + j], hz = gh[base + HH + j], hn  = gh[base + 2 * HH + j];
    float r = 1.0f / (1.0f + expf(-(ir + hr)));
    float z = 1.0f / (1.0f + expf(-(iz + hz)));
    float n = tanhf(in_ + r * hn);
    float hv = (1.0f - z) * n + z * h[i];
    hnew[i]   = hv;
    hnew_h[i] = (_Float16)hv;
}

// scores[s,b] = sum_a v[a] * tanh(h_part[b,a] + enc_part[s,b,a])
__global__ __launch_bounds__(128) void attn_scores_kernel(
    const float* __restrict__ h_part, const float* __restrict__ enc_part,
    const float* __restrict__ v, float* __restrict__ scores)
{
    int sb = blockIdx.x;                           // s*B + b
    int b  = sb & (BB - 1);
    const float* ep = enc_part + (size_t)sb * AAD;
    const float* hp = h_part + (size_t)b * AAD;
    float sum = 0.0f;
    for (int a = threadIdx.x; a < AAD; a += 128)
        sum += v[a] * tanhf(hp[a] + ep[a]);
    __shared__ float red[128];
    red[threadIdx.x] = sum;
    __syncthreads();
    for (int off = 64; off; off >>= 1) {
        if (threadIdx.x < off) red[threadIdx.x] += red[threadIdx.x + off];
        __syncthreads();
    }
    if (threadIdx.x == 0) scores[sb] = red[0];
}

// per-batch softmax over S, emit attn row, ctx = sum_s alpha*enc_ops,
// build f16 [ctx, emb] row for the decoder input GEMM
__global__ __launch_bounds__(128) void softmax_ctx_kernel(
    const float* __restrict__ scores, const float* __restrict__ enc_ops,
    const float* __restrict__ emb_tgt, const int* __restrict__ xcur,
    _Float16* __restrict__ xcat_h, float* __restrict__ attn_out)
{
    int b = blockIdx.x;
    __shared__ float alpha[SS];
    if (threadIdx.x < SS) alpha[threadIdx.x] = scores[threadIdx.x * BB + b];
    __syncthreads();
    if (threadIdx.x == 0) {
        float m = alpha[0];
        for (int s = 1; s < SS; ++s) m = fmaxf(m, alpha[s]);
        float sum = 0.0f;
        for (int s = 0; s < SS; ++s) { float e = expf(alpha[s] - m); alpha[s] = e; sum += e; }
        float inv = 1.0f / sum;
        for (int s = 0; s < SS; ++s) alpha[s] *= inv;
    }
    __syncthreads();
    if (threadIdx.x < SS) attn_out[(size_t)b * SS + threadIdx.x] = alpha[threadIdx.x];

    for (int j = threadIdx.x; j < HH; j += 128) {
        float c = 0.0f;
        for (int s = 0; s < SS; ++s)
            c += alpha[s] * enc_ops[((size_t)s * BB + b) * HH + j];
        xcat_h[(size_t)b * (HH + EE) + j] = (_Float16)c;
    }
    int tok = xcur[b];
    for (int e = threadIdx.x; e < EE; e += 128)
        xcat_h[(size_t)b * (HH + EE) + HH + e] = (_Float16)emb_tgt[(size_t)tok * EE + e];
}

// logits = h_new @ W_out + b_out ; pick next token (teacher forcing or argmax)
__global__ __launch_bounds__(128) void logits_kernel(
    const float* __restrict__ h, const float* __restrict__ W_out,
    const float* __restrict__ b_out, const int* __restrict__ target_t,
    const int* __restrict__ tf, int* __restrict__ xnext, float* __restrict__ out_t)
{
    int b = blockIdx.x;
    __shared__ float hs[HH];
    __shared__ float lg[VTT];
    for (int j = threadIdx.x; j < HH; j += 128) hs[j] = h[(size_t)b * HH + j];
    __syncthreads();
    for (int v = threadIdx.x; v < VTT; v += 128) {
        float s = b_out[v];
        for (int k = 0; k < HH; ++k) s += hs[k] * W_out[(size_t)k * VTT + v];
        lg[v] = s;
        out_t[(size_t)b * VTT + v] = s;
    }
    __syncthreads();
    if (threadIdx.x == 0) {
        int best = 0; float bv = lg[0];
        for (int v = 1; v < VTT; ++v) if (lg[v] > bv) { bv = lg[v]; best = v; }
        xnext[b] = (*tf) ? target_t[b] : best;
    }
}

// ---------------------------------------------------------------------------
// Host driver
// ---------------------------------------------------------------------------
extern "C" void kernel_launch(void* const* d_in, const int* in_sizes, int n_in,
                              void* d_out, int out_size, void* d_ws, size_t ws_size,
                              hipStream_t stream)
{
    const int*   source   = (const int*)  d_in[0];
    const int*   target   = (const int*)  d_in[1];
    const int*   tf       = (const int*)  d_in[2];
    const float* emb_src  = (const float*)d_in[3];
    const float* W_ih_enc = (const float*)d_in[4];
    const float* W_hh_enc = (const float*)d_in[5];
    const float* b_ih_enc = (const float*)d_in[6];
    const float* b_hh_enc = (const float*)d_in[7];
    const float* emb_tgt  = (const float*)d_in[8];
    const float* W_attn   = (const float*)d_in[9];
    const float* b_attn   = (const float*)d_in[10];
    const float* v_attn   = (const float*)d_in[11];
    const float* W_ih_dec = (const float*)d_in[12];
    const float* W_hh_dec = (const float*)d_in[13];
    const float* b_ih_dec = (const float*)d_in[14];
    const float* b_hh_dec = (const float*)d_in[15];
    const float* W_out    = (const float*)d_in[16];
    const float* b_out    = (const float*)d_in[17];

    float* out_logits = (float*)d_out;                               // (T,B,VT)
    float* out_attn   = out_logits + (size_t)TT * BB * VTT;          // (T-1,B,S)

    // ---- workspace carving (256B aligned) ----
    char* wsb = (char*)d_ws;
    size_t off = 0;
    auto alloc = [&](size_t bytes) -> void* {
        void* p = wsb + off;
        off = (off + bytes + 255) & ~(size_t)255;
        return p;
    };
    float* enc_ops   = (float*)alloc((size_t)SS * BB * HH * 4);      // (S,B,H) f32
    float* enc_part  = (float*)alloc((size_t)SS * BB * AAD * 4);     // (S,B,A) f32
    float* gi        = (float*)alloc((size_t)BB * 3 * HH * 4);
    float* gh        = (float*)alloc((size_t)BB * 3 * HH * 4);
    float* h0        = (float*)alloc((size_t)BB * HH * 4);
    float* hbuf0     = (float*)alloc((size_t)BB * HH * 4);
    float* hbuf1     = (float*)alloc((size_t)BB * HH * 4);
    float* h_part    = (float*)alloc((size_t)BB * AAD * 4);
    float* scores    = (float*)alloc((size_t)SS * BB * 4);
    int*   xcur      = (int*)  alloc((size_t)BB * 4);
    _Float16* enc_ops_h = (_Float16*)alloc((size_t)SS * BB * HH * 2); // f16 shadow
    _Float16* h0_h      = (_Float16*)alloc((size_t)BB * HH * 2);
    _Float16* hbuf0_h   = (_Float16*)alloc((size_t)BB * HH * 2);
    _Float16* hbuf1_h   = (_Float16*)alloc((size_t)BB * HH * 2);
    _Float16* xcat_h    = (_Float16*)alloc((size_t)BB * (HH + EE) * 2);
    _Float16* xemb_h    = (_Float16*)alloc((size_t)BB * EE * 2);
    _Float16* Wih_enc_h = (_Float16*)alloc((size_t)3 * HH * EE * 2);
    _Float16* Whh_enc_h = (_Float16*)alloc((size_t)3 * HH * HH * 2);
    _Float16* WattnT_h  = (_Float16*)alloc((size_t)AAD * 2 * HH * 2);
    _Float16* Wih_dec_h = (_Float16*)alloc((size_t)3 * HH * (HH + EE) * 2);
    _Float16* Whh_dec_h = (_Float16*)alloc((size_t)3 * HH * HH * 2);

    const dim3 blk32(32), blk128(128), blk256(256);
    auto grid1 = [](int n) { return dim3((n + 255) / 256); };

    // ---- one-time (per call) prep ----
    cvt_f16_kernel<<<grid1(3 * HH * EE), blk256, 0, stream>>>(W_ih_enc, Wih_enc_h, 3 * HH * EE);
    cvt_f16_kernel<<<grid1(3 * HH * HH), blk256, 0, stream>>>(W_hh_enc, Whh_enc_h, 3 * HH * HH);
    cvt_f16_kernel<<<grid1(3 * HH * (HH + EE)), blk256, 0, stream>>>(W_ih_dec, Wih_dec_h, 3 * HH * (HH + EE));
    cvt_f16_kernel<<<grid1(3 * HH * HH), blk256, 0, stream>>>(W_hh_dec, Whh_dec_h, 3 * HH * HH);
    transpose_cvt_kernel<<<grid1(2 * HH * AAD), blk256, 0, stream>>>(W_attn, WattnT_h);
    zero_f32_kernel<<<grid1(BB * HH), blk256, 0, stream>>>(h0, BB * HH);
    zero_f16_kernel<<<grid1(BB * HH), blk256, 0, stream>>>(h0_h, BB * HH);
    init_row0_kernel<<<grid1(BB * VTT), blk256, 0, stream>>>(out_logits);
    init_xcur_kernel<<<grid1(BB), blk256, 0, stream>>>(target, xcur);

    // ---- encoder: 32 GRU steps ----
    const float*    hprev_enc   = h0;
    const _Float16* hprev_enc_h = h0_h;
    for (int s = 0; s < SS; ++s) {
        gather_src_kernel<<<grid1(BB * EE), blk256, 0, stream>>>(source + s * BB, emb_src, xemb_h);
        // gi = x_emb @ W_ih_enc^T + b_ih_enc   (512 x 3072 x 256)
        gemm_wmma_f16<<<dim3(3 * HH / 64, BB / 32), blk32, 0, stream>>>(
            xemb_h, Wih_enc_h, b_ih_enc, gi, EE, EE, EE, 3 * HH);
        // gh = h @ W_hh_enc^T + b_hh_enc       (512 x 3072 x 1024)
        gemm_wmma_f16<<<dim3(3 * HH / 64, BB / 32), blk32, 0, stream>>>(
            hprev_enc_h, Whh_enc_h, b_hh_enc, gh, HH, HH, HH, 3 * HH);
        float*    hnew   = enc_ops   + (size_t)s * BB * HH;
        _Float16* hnew_h = enc_ops_h + (size_t)s * BB * HH;
        gru_gate_kernel<<<grid1(BB * HH), blk256, 0, stream>>>(gi, gh, hprev_enc, hnew, hnew_h);
        hprev_enc   = hnew;
        hprev_enc_h = hnew_h;
    }

    // ---- attention precompute: enc_part = enc_ops @ W_attn[H:2H]^T + b_attn ----
    // (16384 x 1024 x 1024), W rows are columns H..2H of W_attn_T (ldw = 2H, offset H)
    gemm_wmma_f16<<<dim3(AAD / 64, SS * BB / 32), blk32, 0, stream>>>(
        enc_ops_h, WattnT_h + HH, b_attn, enc_part, HH, HH, 2 * HH, AAD);

    // ---- decoder: 31 steps ----
    const float*    hprev   = enc_ops   + (size_t)(SS - 1) * BB * HH;   // h_enc
    const _Float16* hprev_h = enc_ops_h + (size_t)(SS - 1) * BB * HH;
    for (int t = 1; t < TT; ++t) {
        // h_part = h @ W_attn[:H]^T            (512 x 1024 x 1024)
        gemm_wmma_f16<<<dim3(AAD / 64, BB / 32), blk32, 0, stream>>>(
            hprev_h, WattnT_h, nullptr, h_part, HH, HH, 2 * HH, AAD);
        attn_scores_kernel<<<dim3(SS * BB), blk128, 0, stream>>>(h_part, enc_part, v_attn, scores);
        softmax_ctx_kernel<<<dim3(BB), blk128, 0, stream>>>(
            scores, enc_ops, emb_tgt, xcur, xcat_h, out_attn + (size_t)(t - 1) * BB * SS);
        // gi = [ctx, emb] @ W_ih_dec^T + b_ih_dec   (512 x 3072 x 1280)
        gemm_wmma_f16<<<dim3(3 * HH / 64, BB / 32), blk32, 0, stream>>>(
            xcat_h, Wih_dec_h, b_ih_dec, gi, HH + EE, HH + EE, HH + EE, 3 * HH);
        // gh = h @ W_hh_dec^T + b_hh_dec            (512 x 3072 x 1024)
        gemm_wmma_f16<<<dim3(3 * HH / 64, BB / 32), blk32, 0, stream>>>(
            hprev_h, Whh_dec_h, b_hh_dec, gh, HH, HH, HH, 3 * HH);
        float*    hnew   = (t & 1) ? hbuf0   : hbuf1;
        _Float16* hnew_h = (t & 1) ? hbuf0_h : hbuf1_h;
        gru_gate_kernel<<<grid1(BB * HH), blk256, 0, stream>>>(gi, gh, hprev, hnew, hnew_h);
        logits_kernel<<<dim3(BB), blk128, 0, stream>>>(
            hnew, W_out, b_out, target + t * BB, tf, xcur,
            out_logits + (size_t)t * BB * VTT);
        hprev   = hnew;
        hprev_h = hnew_h;
    }
    (void)in_sizes; (void)n_in; (void)out_size; (void)ws_size;
}
